// MutualRefineAndPooling_88192858456074
// MI455X (gfx1250) — matrix operand order repaired and checked by reading, not compiled, not run on hardware
//
#include <hip/hip_runtime.h>
#include <math.h>

typedef __attribute__((ext_vector_type(16))) _Float16 v16h;
typedef __attribute__((ext_vector_type(8)))  _Float16 v8h;
typedef __attribute__((ext_vector_type(4)))  _Float16 v4h;
typedef __attribute__((ext_vector_type(8)))  float    v8f;
typedef __attribute__((ext_vector_type(4)))  float    v4f;

#define Dm 172
#define Bm 32768
#define Tm 3
#define Wm 5
#define D2 344
#define D3 516
#define D4 688
#define INV_SQRT_D 0.07624928774f   // 1/sqrt(172)
#define DV 43                        // Dm/4 float4 vecs
// padded-to-32 K dimensions
#define DmP  192
#define D2P  352
#define D3P  544
#define D4P  704

// ---------------------------------------------------------------- WMMA GEMM
// C = epilogue( A[M][Kp](f16, zero-padded) @ WT[Na][Kp]^T(f16) + bias + Cin )
// Double-buffered LDS, pipelined global loads, no bounds checks in hot loop.
// Requirements: M%128==0, Kp%32==0, WT rows padded (zeros) to >= grid cover.
#define BMt 128
#define BNt 64
#define BKt 32

static __device__ __forceinline__ v16h combine8(v8h lo, v8h hi) {
  v16h r;
#pragma unroll
  for (int i = 0; i < 8; ++i) { r[i] = lo[i]; r[8 + i] = hi[i]; }
  return r;
}

__global__ __launch_bounds__(256)
void wmma_gemm_pipe(const _Float16* __restrict__ A,   // [M][Kp]
                    const _Float16* __restrict__ WT,  // [Na][Kp] (pre-transposed)
                    const float* __restrict__ bias,   // [N] or null
                    const float* __restrict__ Cin,    // [M][ldc32] or null
                    float* __restrict__ C32, int ldc32,     // or null
                    _Float16* __restrict__ C16, int ldc16,  // or null (row pad = ldc16)
                    int M, int N, int Kp, int relu, int flat3)
{
  __shared__ __align__(32) _Float16 As[2][BMt][BKt];   // 2 x 8 KB
  __shared__ __align__(32) _Float16 Bs[2][BNt][BKt];   // 2 x 4 KB

  const int m0   = blockIdx.x * BMt;
  const int n0   = blockIdx.y * BNt;
  const int tid  = threadIdx.x;
  const int wave = tid >> 5;
  const int lane = tid & 31;
  const int wm   = wave >> 1;          // 4 waves along M
  const int wn   = wave & 1;           // 2 waves along N
  const int r16  = lane & 15;
  const int hi   = lane >> 4;

  // staging: A tile = 512 x 16B chunks (2/thread), B tile = 256 x 16B (1/thread)
  const int ac  = tid * 2;
  const int ar0 = ac >> 2,       ao0 = (ac & 3) * 8;
  const int ar1 = (ac + 1) >> 2, ao1 = ((ac + 1) & 3) * 8;
  const int br  = tid >> 2,      bo  = (tid & 3) * 8;

  const _Float16* Abase = A  + (size_t)m0 * Kp;
  const _Float16* Bbase = WT + (size_t)n0 * Kp;

  v8h ra0, ra1, rb;
  auto loadTile = [&](int k0) {
    ra0 = *(const v8h*)(Abase + (size_t)ar0 * Kp + k0 + ao0);
    ra1 = *(const v8h*)(Abase + (size_t)ar1 * Kp + k0 + ao1);
    rb  = *(const v8h*)(Bbase + (size_t)br  * Kp + k0 + bo);
  };
  auto storeTile = [&](int buf) {
    *(v8h*)&As[buf][ar0][ao0] = ra0;
    *(v8h*)&As[buf][ar1][ao1] = ra1;
    *(v8h*)&Bs[buf][br][bo]   = rb;
  };

  v8f acc[2][2] = {};

  loadTile(0);
  storeTile(0);
  __syncthreads();

  const int nk = Kp >> 5;
  for (int kt = 0; kt < nk; ++kt) {
    const int cur = kt & 1;
    const bool hasNext = (kt + 1) < nk;
    if (hasNext) {
      loadTile((kt + 1) << 5);                       // in flight under the WMMAs
      __builtin_prefetch(Abase + (size_t)ar0 * Kp + ((kt + 2) << 5) + ao0, 0, 3);
      __builtin_prefetch(Bbase + (size_t)br  * Kp + ((kt + 2) << 5) + bo,  0, 3);
    }

    v16h af[2], bf[2];
#pragma unroll
    for (int t = 0; t < 2; ++t) {
      int arow = wm * 32 + t * 16 + r16;
      v8h alo = *(const v8h*)&As[cur][arow][hi * 16];
      v8h ahi = *(const v8h*)&As[cur][arow][hi * 16 + 8];
      af[t] = combine8(alo, ahi);
      int bc = wn * 32 + t * 16 + r16;
      bf[t] = *(const v16h*)&Bs[cur][bc][hi * 16];
    }
#pragma unroll
    for (int i = 0; i < 2; ++i)
#pragma unroll
      for (int j = 0; j < 2; ++j)
        acc[i][j] = __builtin_amdgcn_wmma_f32_16x16x32_f16(
            false, af[i], false, bf[j], (short)0, acc[i][j], false, false);

    if (hasNext) storeTile(cur ^ 1);
    __syncthreads();
  }

  // ---- epilogue: lane = N col, vgpr e -> M row = e + hi*8 (M never OOB)
#pragma unroll
  for (int i = 0; i < 2; ++i)
#pragma unroll
    for (int j = 0; j < 2; ++j) {
      int gn  = n0 + wn * 32 + j * 16 + r16;
      int gm0 = m0 + wm * 32 + i * 16 + hi * 8;
      float bv = (bias && gn < N) ? bias[gn] : 0.0f;
#pragma unroll
      for (int e = 0; e < 8; ++e) {
        int gm = gm0 + e;
        float v = acc[i][j][e];
        if (gn < N) {
          v += bv;
          if (Cin) v += Cin[(size_t)gm * ldc32 + gn];
          if (relu) v = fmaxf(v, 0.0f);
          if (C32) C32[(size_t)gm * ldc32 + gn] = v;
        }
        if (C16) {
          if (flat3) {           // [B*T, N] -> [B, 3N] with row pad ldc16
            if (gn < N)
              C16[(size_t)(gm / 3) * ldc16 + (gm % 3) * N + gn] = (_Float16)v;
          } else if (gn < ldc16) {
            C16[(size_t)gm * ldc16 + gn] = (_Float16)(gn < N ? v : 0.0f);
          }
        }
      }
    }
}

// ------------------------------------------- weight transpose f32 -> f16 pad
// WT[n][k] = (k<K && n<N) ? W[k*lds + n] : 0 ; for all n<Na, k<Kp
__global__ __launch_bounds__(256)
void transpose_w_kernel(const float* __restrict__ W, int lds_, int K, int N,
                        _Float16* __restrict__ WT, int Kp, int Na)
{
  __shared__ float tile[32][33];
  int tx = threadIdx.x & 31;
  int ty = threadIdx.x >> 5;                 // 0..7
  int k0 = blockIdx.x * 32;
  int n0 = blockIdx.y * 32;
#pragma unroll
  for (int j = 0; j < 4; ++j) {
    int k = k0 + ty + j * 8;
    int n = n0 + tx;
    tile[ty + j * 8][tx] = (k < K && n < N) ? W[(size_t)k * lds_ + n] : 0.0f;
  }
  __syncthreads();
#pragma unroll
  for (int j = 0; j < 4; ++j) {
    int n = n0 + ty + j * 8;
    int k = k0 + tx;
    if (n < Na && k < Kp)
      WT[(size_t)n * Kp + k] = (_Float16)tile[tx][ty + j * 8];
  }
}

// ------------------------------------------- f32 -> padded f16 row convert
__global__ __launch_bounds__(256)
void cvt_pad_kernel(const float* __restrict__ X, int N,
                    _Float16* __restrict__ Y, int Np, long rows)
{
  long idx = (long)blockIdx.x * 256 + threadIdx.x;
  if (idx >= rows * Np) return;
  int  n = (int)(idx % Np);
  long r = idx / Np;
  Y[idx] = (_Float16)((n < N) ? X[r * N + n] : 0.0f);
}

// zero the [516,544) pad columns of flat16
__global__ void pad_flat_kernel(_Float16* __restrict__ flat16)
{
  int r = blockIdx.x, t = threadIdx.x;
  if (t < D3P - D3) flat16[(size_t)r * D3P + D3 + t] = (_Float16)0.0f;
}

// ------------------------------------------------- pooling vector precompute
__global__ void pool_vec_kernel(const float* __restrict__ pq,
                                const float* __restrict__ w_in_p,
                                const float* __restrict__ b_in_p,
                                float* __restrict__ kq)
{
  __shared__ float qs[Dm];
  int t = threadIdx.x;
  if (t < Dm) {
    float acc = b_in_p[t];
    for (int d = 0; d < Dm; ++d) acc += pq[d] * w_in_p[(size_t)d * D3 + t];
    qs[t] = acc;
  }
  __syncthreads();
  if (t < Dm) {
    float acc = 0.0f;
    for (int e = 0; e < Dm; ++e) acc += w_in_p[(size_t)t * D3 + Dm + e] * qs[e];
    kq[t] = acc * INV_SQRT_D;
  }
  if (t == 0) {
    float acc = 0.0f;
    for (int e = 0; e < Dm; ++e) acc += b_in_p[Dm + e] * qs[e];
    kq[Dm] = acc * INV_SQRT_D;
  }
}

// ------------------------------------- fused score/softmax/weighted-sum pool
// one wave per (b,t); per_type read exactly once (NT stream); emits padded f16
__global__ __launch_bounds__(256)
void pool_attn_kernel(const float* __restrict__ X,           // [rows, Wm, Dm]
                      const unsigned char* __restrict__ mask,// [rows, Wm]
                      const float* __restrict__ kq,          // [Dm+1]
                      _Float16* __restrict__ xa16,           // [rows, DmP]
                      int rows)
{
  int wave = threadIdx.x >> 5, lane = threadIdx.x & 31;
  long row = (long)blockIdx.x * 8 + wave;
  if (row >= rows) return;
  const v4f* xr4 = (const v4f*)(X + row * (Wm * Dm));
  const v4f* kq4 = (const v4f*)kq;

  v4f xv[Wm][2];
  v4f kv[2];
#pragma unroll
  for (int i = 0; i < 2; ++i) {
    int v = lane + i * 32;
    bool ok = v < DV;
    int vc = ok ? v : 0;
    kv[i] = kq4[vc];
    if (!ok) kv[i] = (v4f){0.f, 0.f, 0.f, 0.f};
#pragma unroll
    for (int w = 0; w < Wm; ++w) {
      v4f x = __builtin_nontemporal_load(&xr4[w * DV + vc]);
      xv[w][i] = ok ? x : (v4f){0.f, 0.f, 0.f, 0.f};
    }
  }

  float kqb = kq[Dm];
  float s[Wm];
#pragma unroll
  for (int w = 0; w < Wm; ++w) {
    float p = 0.0f;
#pragma unroll
    for (int i = 0; i < 2; ++i) {
      v4f x = xv[w][i], k = kv[i];
      p += x[0]*k[0] + x[1]*k[1] + x[2]*k[2] + x[3]*k[3];
    }
#pragma unroll
    for (int off = 16; off; off >>= 1) p += __shfl_xor(p, off, 32);
    s[w] = p + kqb;
  }
  float mx = -1e30f;
#pragma unroll
  for (int w = 0; w < Wm; ++w) {
    if (!mask[row * Wm + w]) s[w] = -1e9f;
    mx = fmaxf(mx, s[w]);
  }
  float sum = 0.0f;
#pragma unroll
  for (int w = 0; w < Wm; ++w) { s[w] = __expf(s[w] - mx); sum += s[w]; }
  float inv = 1.0f / sum;

  _Float16* xo = xa16 + row * DmP;
#pragma unroll
  for (int i = 0; i < 2; ++i) {
    int v = lane + i * 32;
    if (v >= DmP / 4) continue;               // 48 vec4 slots per row
    v4h o = {};
    if (v < DV) {
      v4f a = (v4f){0.f, 0.f, 0.f, 0.f};
#pragma unroll
      for (int w = 0; w < Wm; ++w) {
        v4f x = xv[w][i];
        a[0] += s[w]*x[0]; a[1] += s[w]*x[1]; a[2] += s[w]*x[2]; a[3] += s[w]*x[3];
      }
#pragma unroll
      for (int j = 0; j < 4; ++j) o[j] = (_Float16)(a[j] * inv);
    }
    *(v4h*)(xo + v * 4) = o;
  }
}

// ------------------------------------------------ sigmoid gate + residual
__global__ __launch_bounds__(256)
void gate_residual_kernel(const float* __restrict__ walk,
                          const float* __restrict__ gatepre,
                          const float* __restrict__ cross,
                          float* __restrict__ refined, long n4)
{
  long i = (long)blockIdx.x * 256 + threadIdx.x;
  if (i >= n4) return;
  v4f w = ((const v4f*)walk)[i];
  v4f g = ((const v4f*)gatepre)[i];
  v4f c = ((const v4f*)cross)[i];
  v4f r;
#pragma unroll
  for (int j = 0; j < 4; ++j)
    r[j] = w[j] + c[j] / (1.0f + __expf(-g[j]));
  ((v4f*)refined)[i] = r;
}

// ---------------------------------------------------------------- LayerNorm
__global__ __launch_bounds__(256)
void layernorm_kernel(const float* __restrict__ x,
                      const float* __restrict__ g,
                      const float* __restrict__ b,
                      float* __restrict__ out, long rows)
{
  int wave = threadIdx.x >> 5, lane = threadIdx.x & 31;
  long row = (long)blockIdx.x * 8 + wave;
  if (row >= rows) return;
  const v4f* xr4 = (const v4f*)(x + row * Dm);
  v4f xv[2];
  float s = 0.0f, s2 = 0.0f;
#pragma unroll
  for (int i = 0; i < 2; ++i) {
    int v = lane + i * 32;
    bool ok = v < DV;
    v4f xx = xr4[ok ? v : 0];
    if (!ok) xx = (v4f){0.f, 0.f, 0.f, 0.f};
    xv[i] = xx;
#pragma unroll
    for (int j = 0; j < 4; ++j) { s += xx[j]; s2 += xx[j] * xx[j]; }
  }
#pragma unroll
  for (int off = 16; off; off >>= 1) {
    s  += __shfl_xor(s,  off, 32);
    s2 += __shfl_xor(s2, off, 32);
  }
  float mean = s * (1.0f / Dm);
  float var  = s2 * (1.0f / Dm) - mean * mean;
  float rstd = rsqrtf(var + 1e-5f);
  v4f* o4 = (v4f*)(out + row * Dm);
  const v4f* g4 = (const v4f*)g;
  const v4f* b4 = (const v4f*)b;
#pragma unroll
  for (int i = 0; i < 2; ++i) {
    int v = lane + i * 32;
    if (v >= DV) continue;
    v4f gg = g4[v], bb = b4[v], xx = xv[i], r;
#pragma unroll
    for (int j = 0; j < 4; ++j) r[j] = (xx[j] - mean) * rstd * gg[j] + bb[j];
    o4[v] = r;
  }
}

// ---------------------------------------------------------------------------
static inline void launch_gemm(hipStream_t s,
                               const _Float16* A, const _Float16* WT,
                               const float* bias, const float* Cin,
                               float* C32, int ldc32,
                               _Float16* C16, int ldc16,
                               int M, int N, int Kp, int relu, int flat3)
{
  dim3 grid(M / BMt, (N + BNt - 1) / BNt);
  hipLaunchKernelGGL(wmma_gemm_pipe, grid, dim3(256), 0, s,
                     A, WT, bias, Cin, C32, ldc32, C16, ldc16, M, N, Kp, relu, flat3);
}

static inline void launch_transpose(hipStream_t s, const float* W, int lds_,
                                    int K, int N, _Float16* WT, int Kp, int Na)
{
  dim3 grid((Kp + 31) / 32, (Na + 31) / 32);
  hipLaunchKernelGGL(transpose_w_kernel, grid, dim3(256), 0, s,
                     W, lds_, K, N, WT, Kp, Na);
}

extern "C" void kernel_launch(void* const* d_in, const int* in_sizes, int n_in,
                              void* d_out, int out_size, void* d_ws, size_t ws_size,
                              hipStream_t stream)
{
  const float* src_walk  = (const float*)d_in[0];
  const float* dst_walk  = (const float*)d_in[1];
  const float* src_pt    = (const float*)d_in[2];
  const float* dst_pt    = (const float*)d_in[3];
  const unsigned char* src_mask = (const unsigned char*)d_in[4];
  const unsigned char* dst_mask = (const unsigned char*)d_in[5];
  const float* pool_query = (const float*)d_in[6];
  const float* w_in_s2d  = (const float*)d_in[7];
  const float* b_in_s2d  = (const float*)d_in[8];
  const float* w_out_s2d = (const float*)d_in[9];
  const float* b_out_s2d = (const float*)d_in[10];
  const float* w_in_d2s  = (const float*)d_in[11];
  const float* b_in_d2s  = (const float*)d_in[12];
  const float* w_out_d2s = (const float*)d_in[13];
  const float* b_out_d2s = (const float*)d_in[14];
  const float* w_in_p    = (const float*)d_in[15];
  const float* b_in_p    = (const float*)d_in[16];
  const float* w_out_p   = (const float*)d_in[17];
  const float* b_out_p   = (const float*)d_in[18];
  const float* w_gs = (const float*)d_in[19]; const float* b_gs = (const float*)d_in[20];
  const float* w_gd = (const float*)d_in[21]; const float* b_gd = (const float*)d_in[22];
  const float* w_fs1 = (const float*)d_in[23]; const float* b_fs1 = (const float*)d_in[24];
  const float* w_fs2 = (const float*)d_in[25]; const float* b_fs2 = (const float*)d_in[26];
  const float* w_fd1 = (const float*)d_in[27]; const float* b_fd1 = (const float*)d_in[28];
  const float* w_fd2 = (const float*)d_in[29]; const float* b_fd2 = (const float*)d_in[30];
  const float* g_s  = (const float*)d_in[31]; const float* be_s = (const float*)d_in[32];
  const float* g_d  = (const float*)d_in[33]; const float* be_d = (const float*)d_in[34];
  const float* w_ffs1 = (const float*)d_in[35]; const float* b_ffs1 = (const float*)d_in[36];
  const float* w_ffs2 = (const float*)d_in[37]; const float* b_ffs2 = (const float*)d_in[38];
  const float* w_ffd1 = (const float*)d_in[39]; const float* b_ffd1 = (const float*)d_in[40];
  const float* w_ffd2 = (const float*)d_in[41]; const float* b_ffd2 = (const float*)d_in[42];
  (void)in_sizes; (void)n_in; (void)out_size; (void)ws_size;

  // ------------------------------------------------ workspace layout
  float* kq = (float*)d_ws;                                // 512 floats
  float* R1 = kq + 512;                                    // gatepre  [B,D] f32
  float* R2 = R1 + (size_t)Bm * Dm;                        // cross    [B,D] f32
  float* R3 = R2 + (size_t)Bm * Dm;                        // refined  [B,D] f32
  _Float16* walkS16 = (_Float16*)(R3 + (size_t)Bm * Dm);   // [B,192]
  _Float16* walkD16 = walkS16 + (size_t)Bm * DmP;          // [B,192]
  _Float16* PoolA   = walkD16 + (size_t)Bm * DmP;          // Bm*576 halves
  _Float16* PoolB   = PoolA + (size_t)Bm * (Tm * DmP);     // Bm*704 halves
  _Float16* wt      = PoolB + (size_t)Bm * D4P;
  _Float16* WT_pV   = wt;                 wt += DmP * DmP;
  _Float16* WT_pOut = wt;                 wt += DmP * DmP;
  _Float16* WT_inV  = wt;                 wt += DmP * DmP;
  _Float16* WT_out  = wt;                 wt += DmP * DmP;
  _Float16* WT_gt   = wt;                 wt += DmP * DmP;
  _Float16* WT_gb   = wt;                 wt += DmP * DmP;
  _Float16* WT_f1   = wt;                 wt += 384 * D3P;
  _Float16* WT_f2   = wt;                 wt += DmP * D2P;
  _Float16* WT_ff1  = wt;                 wt += D4P * DmP;
  _Float16* WT_ff2  = wt;                 wt += DmP * D4P;

  float* out = (float*)d_out;

  // pooling q/kq + shared pooling weights (both sides)
  hipLaunchKernelGGL(pool_vec_kernel, dim3(1), dim3(192), 0, stream,
                     pool_query, w_in_p, b_in_p, kq);
  launch_transpose(stream, w_in_p + D2, D3, Dm, Dm, WT_pV, DmP, DmP);
  launch_transpose(stream, w_out_p,     Dm, Dm, Dm, WT_pOut, DmP, DmP);

  // walk inputs -> padded f16 (used as GEMM A)
  {
    long tot = (long)Bm * DmP;
    unsigned blocks = (unsigned)((tot + 255) / 256);
    hipLaunchKernelGGL(cvt_pad_kernel, dim3(blocks), dim3(256), 0, stream,
                       src_walk, Dm, walkS16, DmP, (long)Bm);
    hipLaunchKernelGGL(cvt_pad_kernel, dim3(blocks), dim3(256), 0, stream,
                       dst_walk, Dm, walkD16, DmP, (long)Bm);
  }

  const float* walk_self[2]  = {src_walk, dst_walk};
  const _Float16* walk16_self[2]  = {walkS16, walkD16};
  const _Float16* walk16_other[2] = {walkD16, walkS16};
  const float* pt[2]    = {src_pt, dst_pt};
  const unsigned char* msk[2] = {src_mask, dst_mask};
  const float* w_in[2]  = {w_in_s2d, w_in_d2s};
  const float* b_in[2]  = {b_in_s2d, b_in_d2s};
  const float* w_o[2]   = {w_out_s2d, w_out_d2s};
  const float* b_o[2]   = {b_out_s2d, b_out_d2s};
  const float* w_g[2]   = {w_gs, w_gd};
  const float* b_g[2]   = {b_gs, b_gd};
  const float* w_f1[2]  = {w_fs1, w_fd1};
  const float* b_f1[2]  = {b_fs1, b_fd1};
  const float* w_f2[2]  = {w_fs2, w_fd2};
  const float* b_f2[2]  = {b_fs2, b_fd2};
  const float* w_ff1[2] = {w_ffs1, w_ffd1};
  const float* b_ff1[2] = {b_ffs1, b_ffd1};
  const float* w_ff2[2] = {w_ffs2, w_ffd2};
  const float* b_ff2[2] = {b_ffs2, b_ffd2};
  const float* lng[2]   = {g_s, g_d};
  const float* lnb[2]   = {be_s, be_d};

  const long nBD4 = ((long)Bm * Dm) / 4;
  const int rowsBT = Bm * Tm;

  for (int s = 0; s < 2; ++s) {
    // ---- per-side weight transposes (f16, padded, pre-transposed)
    launch_transpose(stream, w_in[s] + D2, D3, Dm, Dm, WT_inV, DmP, DmP);
    launch_transpose(stream, w_o[s],       Dm, Dm, Dm, WT_out, DmP, DmP);
    launch_transpose(stream, w_g[s],                  Dm, Dm, Dm, WT_gt, DmP, DmP);
    launch_transpose(stream, w_g[s] + (size_t)Dm*Dm,  Dm, Dm, Dm, WT_gb, DmP, DmP);
    launch_transpose(stream, w_f1[s],  D2, D3, D2, WT_f1,  D3P, 384);
    launch_transpose(stream, w_f2[s],  Dm, D2, Dm, WT_f2,  D2P, DmP);
    launch_transpose(stream, w_ff1[s], D4, Dm, D4, WT_ff1, DmP, D4P);
    launch_transpose(stream, w_ff2[s], Dm, D4, Dm, WT_ff2, D4P, DmP);

    _Float16* tmp1_16   = PoolA;   // [B,192]
    _Float16* cross16   = PoolB;   // [B,192]
    _Float16* xa16      = PoolA;   // [B*T,192]
    _Float16* pooled16  = PoolB;   // [B*T,192]
    _Float16* flat16    = PoolA;   // [B,544]
    _Float16* h16       = PoolB;   // [B,352]
    _Float16* refined16 = PoolA;   // [B,192]
    _Float16* h4_16     = PoolB;   // [B,704]

    // 1. tmp1 = walk_other @ V_in + bv
    launch_gemm(stream, walk16_other[s], WT_inV, b_in[s] + D2, nullptr,
                nullptr, 0, tmp1_16, DmP, Bm, Dm, DmP, 0, 0);
    // 2. cross = tmp1 @ Wout + bout   (dual: f32 for gating, f16 for GEMM 4)
    launch_gemm(stream, tmp1_16, WT_out, b_o[s], nullptr,
                R2, Dm, cross16, DmP, Bm, Dm, DmP, 0, 0);
    // 3. gatepre = walk_self @ Wg_top + bg
    launch_gemm(stream, walk16_self[s], WT_gt, b_g[s], nullptr,
                R1, Dm, nullptr, 0, Bm, Dm, DmP, 0, 0);
    // 4. gatepre += cross @ Wg_bot
    launch_gemm(stream, cross16, WT_gb, nullptr, R1,
                R1, Dm, nullptr, 0, Bm, Dm, DmP, 0, 0);
    // 5. refined = walk + sigmoid(gatepre) * cross
    hipLaunchKernelGGL(gate_residual_kernel,
                       dim3((unsigned)((nBD4 + 255) / 256)), dim3(256), 0, stream,
                       walk_self[s], R1, R2, R3, nBD4);
    // 6. fused pooling -> xa16 (one NT pass over per_type)
    hipLaunchKernelGGL(pool_attn_kernel,
                       dim3((rowsBT + 7) / 8), dim3(256), 0, stream,
                       pt[s], msk[s], kq, xa16, rowsBT);
    // 7. pooled = xa @ Vp + bvp
    launch_gemm(stream, xa16, WT_pV, b_in_p + D2, nullptr,
                nullptr, 0, pooled16, DmP, rowsBT, Dm, DmP, 0, 0);
    // 8. flat = pooled @ Wout_p + bout_p  ([B*T,D] -> [B,3D] padded to 544)
    launch_gemm(stream, pooled16, WT_pOut, b_out_p, nullptr,
                nullptr, 0, flat16, D3P, rowsBT, Dm, DmP, 0, 1);
    hipLaunchKernelGGL(pad_flat_kernel, dim3(Bm), dim3(32), 0, stream, flat16);
    // 9. h = relu(flat @ Wf1 + bf1)
    launch_gemm(stream, flat16, WT_f1, b_f1[s], nullptr,
                nullptr, 0, h16, D2P, Bm, D2, D3P, 1, 0);
    // 10. refined += h @ Wf2 + bf2  (dual: f32 chain + f16 for GEMM 11)
    launch_gemm(stream, h16, WT_f2, b_f2[s], R3,
                R3, Dm, refined16, DmP, Bm, Dm, D2P, 0, 0);
    // 11. h4 = relu(refined @ Wff1 + bff1)
    launch_gemm(stream, refined16, WT_ff1, b_ff1[s], nullptr,
                nullptr, 0, h4_16, D4P, Bm, D4, DmP, 1, 0);
    // 12. refined += h4 @ Wff2 + bff2 (in place f32)
    launch_gemm(stream, h4_16, WT_ff2, b_ff2[s], R3,
                R3, Dm, nullptr, 0, Bm, Dm, D4P, 0, 0);
    // 13. out = LayerNorm(refined)
    hipLaunchKernelGGL(layernorm_kernel,
                       dim3((Bm + 7) / 8), dim3(256), 0, stream,
                       R3, lng[s], lnb[s], out + (size_t)s * Bm * Dm, (long)Bm);
  }
}
